// MSRG_10496900071815
// MI455X (gfx1250) — matrix operand order repaired and checked by reading, not compiled, not run on hardware
//
#include <hip/hip_runtime.h>
#include <hip/hip_bf16.h>

// ---------------------------------------------------------------------------
// MSRG multimodal RNN for MI455X (gfx1250, wave32, WMMA).
// All dense math runs through v_wmma_f32_16x16x32_f16 (f16 in, f32 acc).
// All GEMMs have K % 32 == 0 (ragged inputs are zero-padded in workspace),
// so the WMMA inner loop is branch-free with b128 loads.
// ---------------------------------------------------------------------------

typedef __attribute__((ext_vector_type(16))) _Float16 v16h;
typedef __attribute__((ext_vector_type(8)))  _Float16 v8h;
typedef __attribute__((ext_vector_type(8)))  float    v8f;

#define WMMA_F16(a, b, c) \
  __builtin_amdgcn_wmma_f32_16x16x32_f16(false, (a), false, (b), (short)0, (c), false, false)

// ---------------------------------------------------------------------------
// Weight conversion f32 -> f16, output N x Kp row-major, zero-padded K..Kp.
// ---------------------------------------------------------------------------
__global__ void k_cvt_pad(const float* __restrict__ src, _Float16* __restrict__ dst,
                          int rows, int cols, int colsp) {
  int i = blockIdx.x * blockDim.x + threadIdx.x;
  if (i >= rows * colsp) return;
  int r = i / colsp, c = i % colsp;
  dst[i] = (c < cols) ? (_Float16)src[(size_t)r * cols + c] : (_Float16)0.0f;
}

// Transposed conversion: src (K x N) -> dst (N x K), K already mult of 32.
__global__ void k_cvtT(const float* __restrict__ src, _Float16* __restrict__ dst,
                       int rows, int cols) {   // rows = K, cols = N
  int i = blockIdx.x * blockDim.x + threadIdx.x;
  if (i >= rows * cols) return;
  int r = i / cols, c = i % cols;
  dst[(size_t)c * rows + r] = (_Float16)src[i];
}

// Pack a strided f32 slice (M x K, row stride lds) into dense (M x Kp), zero pad.
__global__ void k_pack(const float* __restrict__ src, int lds, int K,
                       float* __restrict__ dst, int Kp, int M) {
  int i = blockIdx.x * blockDim.x + threadIdx.x;
  if (i >= M * Kp) return;
  int r = i / Kp, c = i % Kp;
  dst[i] = (c < K) ? src[(size_t)r * lds + c] : 0.0f;
}

// ---------------------------------------------------------------------------
// C(MxN,f32) = A(MxK,f32,lda) @ W(NxK,f16)^T + bias(N).
// One wave computes NT consecutive 16x16 C tiles in N, reusing the A fragment
// (loaded + cvt'd once per k-step) across NT WMMAs.
// REQUIRES: K % 32 == 0, lda % 4 == 0, A/W 16-byte aligned,
//           M % 16 == 0, N % (16*NT) == 0. Branch-free inner loop.
// Fragment layouts used here (A and B share one contiguous k order):
//   A 16x32 f16 : m = lane&15; half = lane>>4; K = half*16 + {0..15} (contiguous A row)
//   B 32x16 f16 : n = lane&15; K = (lane>>4)*16 + {0..15} (contiguous W row)
//   C 16x16 f32 : n = lane&15; row m = (lane>>4)*8 + j
// ---------------------------------------------------------------------------
template <int NT>
__global__ __launch_bounds__(256)
void k_gemm_aw(const float* __restrict__ A, int lda,
               const _Float16* __restrict__ W,
               const float* __restrict__ bias,
               float* __restrict__ C, int ldc,
               int M, int N, int K) {
  int wave = threadIdx.x >> 5;
  int lane = threadIdx.x & 31;
  int tgn    = (N >> 4) / NT;                // tile-groups along N
  int groups = (M >> 4) * tgn;
  int g = blockIdx.x * 8 + wave;             // wave-uniform: EXEC stays full
  if (g >= groups) return;
  int tm = g / tgn, tg = g % tgn;
  int half = lane >> 4;
  int m  = tm * 16 + (lane & 15);
  int n0 = tg * (16 * NT) + (lane & 15);
  const float* Arow = A + (size_t)m * lda + half * 16;          // 32B aligned
  const _Float16* Wrow[NT];
#pragma unroll
  for (int t = 0; t < NT; ++t)
    Wrow[t] = W + (size_t)(n0 + 16 * t) * K + half * 16;        // 32B aligned

  v8f acc[NT];
#pragma unroll
  for (int t = 0; t < NT; ++t) acc[t] = (v8f){};

  int ksteps = K >> 5;
  for (int kk = 0; kk < ksteps; ++kk) {
    int k0 = kk * 32;
    __builtin_prefetch(Arow + k0 + 64, 0, 1);   // global_prefetch next A chunk
    float4 x0 = *(const float4*)(Arow + k0);
    float4 x1 = *(const float4*)(Arow + k0 + 4);
    float4 x2 = *(const float4*)(Arow + k0 + 8);
    float4 x3 = *(const float4*)(Arow + k0 + 12);
    v16h af;
    af[0]  = (_Float16)x0.x; af[1]  = (_Float16)x0.y;
    af[2]  = (_Float16)x0.z; af[3]  = (_Float16)x0.w;
    af[4]  = (_Float16)x1.x; af[5]  = (_Float16)x1.y;
    af[6]  = (_Float16)x1.z; af[7]  = (_Float16)x1.w;
    af[8]  = (_Float16)x2.x; af[9]  = (_Float16)x2.y;
    af[10] = (_Float16)x2.z; af[11] = (_Float16)x2.w;
    af[12] = (_Float16)x3.x; af[13] = (_Float16)x3.y;
    af[14] = (_Float16)x3.z; af[15] = (_Float16)x3.w;
#pragma unroll
    for (int t = 0; t < NT; ++t) {
      v16h bf = *(const v16h*)(Wrow[t] + k0);   // 32B load, 2 x b128
      acc[t] = WMMA_F16(af, bf, acc[t]);
    }
  }
#pragma unroll
  for (int t = 0; t < NT; ++t) {
    int n = n0 + 16 * t;
    float bn = bias ? bias[n] : 0.0f;
#pragma unroll
    for (int j = 0; j < 8; ++j) {
      int mr = tm * 16 + half * 8 + j;
      C[(size_t)mr * ldc + n] = acc[t][j] + bn;
    }
  }
}

// ---------------------------------------------------------------------------
// GRU gate update. gi_t includes bih; gh includes bhh. h in place; optional
// scatter of h_new into a sequence buffer.
// ---------------------------------------------------------------------------
__global__ void k_gru_gate(const float* __restrict__ gi_t, int gi_rs,
                           const float* __restrict__ gh,
                           float* __restrict__ h,
                           float* __restrict__ out, long out_bs,
                           int B, int H) {
  int idx = blockIdx.x * blockDim.x + threadIdx.x;
  if (idx >= B * H) return;
  int b = idx / H, c = idx % H;
  const float* gib = gi_t + (size_t)b * gi_rs;
  const float* ghb = gh + (size_t)b * 3 * H;
  float r = 1.0f / (1.0f + __expf(-(gib[c] + ghb[c])));
  float z = 1.0f / (1.0f + __expf(-(gib[H + c] + ghb[H + c])));
  float nn = tanhf(gib[2 * H + c] + r * ghb[2 * H + c]);
  float hn = (1.0f - z) * nn + z * h[idx];
  h[idx] = hn;
  if (out) out[(size_t)b * out_bs + c] = hn;
}

// ---------------------------------------------------------------------------
// LSTM gate update (gate order i, f, g, o).
// ---------------------------------------------------------------------------
__global__ void k_lstm_gate(const float* __restrict__ gi_t, int gi_rs,
                            const float* __restrict__ gh,
                            float* __restrict__ h, float* __restrict__ cs,
                            float* __restrict__ out, long out_bs,
                            int B, int H) {
  int idx = blockIdx.x * blockDim.x + threadIdx.x;
  if (idx >= B * H) return;
  int b = idx / H, c = idx % H;
  const float* gib = gi_t + (size_t)b * gi_rs;
  const float* ghb = gh + (size_t)b * 4 * H;
  float ig = 1.0f / (1.0f + __expf(-(gib[c] + ghb[c])));
  float fg = 1.0f / (1.0f + __expf(-(gib[H + c] + ghb[H + c])));
  float gg = tanhf(gib[2 * H + c] + ghb[2 * H + c]);
  float og = 1.0f / (1.0f + __expf(-(gib[3 * H + c] + ghb[3 * H + c])));
  float cn = fg * cs[idx] + ig * gg;
  cs[idx] = cn;
  float hn = og * tanhf(cn);
  h[idx] = hn;
  out[(size_t)b * out_bs + c] = hn;
}

// ---------------------------------------------------------------------------
// Fused per-batch cross-attention: one workgroup (8 waves) per batch.
// Q,K staged as f16 in LDS (Q pre-scaled by 1/sqrt(d)), S = QK^T in f32 LDS,
// softmax, P(f16) overwrites Q, V^T overwrites K, O = PV. T = d = 128.
// A fragments are hoisted into registers across the 8-tile loop; V is stored
// transposed so both matmul stages use contiguous 16B LDS vector loads.
// Dynamic LDS = 2*T*ALD*2 + T*SLD*4 = 137216 bytes (<320 KB/WGP).
// ---------------------------------------------------------------------------
#define ALD 136   // f16 row stride; 136*2 = 272 bytes, 16B-aligned rows
#define SLD 132   // f32 row stride
__global__ __launch_bounds__(256)
void k_attn(const float* __restrict__ q, const float* __restrict__ k,
            const float* __restrict__ v, float* __restrict__ out,
            long out_bstride, int out_rs) {
  const int T = 128, d = 128;
  extern __shared__ char smem[];
  _Float16* Qs = (_Float16*)smem;            // [T][ALD]  (later holds P)
  _Float16* Ks = Qs + T * ALD;               // [T][ALD]  (later holds V^T)
  float*    S  = (float*)(Ks + T * ALD);     // [T][SLD]
  int b = blockIdx.x;
  int tid = threadIdx.x;
  const float scale = rsqrtf((float)d);

  {
    const float4* q4 = (const float4*)(q + (size_t)b * T * d);
    const float4* k4 = (const float4*)(k + (size_t)b * T * d);
    for (int i = tid; i < T * d / 4; i += blockDim.x) {
      int t = i / (d / 4), c = (i % (d / 4)) * 4;
      float4 xq = q4[i];
      float4 xk = k4[i];
      _Float16* dq = Qs + t * ALD + c;
      _Float16* dk = Ks + t * ALD + c;
      dq[0] = (_Float16)(xq.x * scale); dq[1] = (_Float16)(xq.y * scale);
      dq[2] = (_Float16)(xq.z * scale); dq[3] = (_Float16)(xq.w * scale);
      dk[0] = (_Float16)xk.x; dk[1] = (_Float16)xk.y;
      dk[2] = (_Float16)xk.z; dk[3] = (_Float16)xk.w;
    }
  }
  __syncthreads();

  int wave = tid >> 5, lane = tid & 31, half = lane >> 4;
  int mrow = wave * 16 + (lane & 15);

  // S = (Q*scale) @ K^T : hoist the 4 A fragments, then 8 output tiles.
  {
    v16h afq[4];
#pragma unroll
    for (int kk = 0; kk < 4; ++kk) {
      int k0 = kk * 32;
      v8h alo = *(const v8h*)(Qs + mrow * ALD + k0 + half * 8);
      v8h ahi = *(const v8h*)(Qs + mrow * ALD + k0 + 16 + half * 8);
      afq[kk] = __builtin_shufflevector(alo, ahi, 0,1,2,3,4,5,6,7,8,9,10,11,12,13,14,15);
    }
    for (int tn = 0; tn < T / 16; ++tn) {
      int n = tn * 16 + (lane & 15);
      v8f acc = {};
#pragma unroll
      for (int kk = 0; kk < 4; ++kk) {
        int k0 = kk * 32;
        v8h blo = *(const v8h*)(Ks + n * ALD + k0 + half * 8);
        v8h bhi = *(const v8h*)(Ks + n * ALD + k0 + 16 + half * 8);
        v16h bf = __builtin_shufflevector(blo, bhi, 0,1,2,3,4,5,6,7,8,9,10,11,12,13,14,15);
        acc = WMMA_F16(afq[kk], bf, acc);
      }
#pragma unroll
      for (int j = 0; j < 8; ++j)
        S[(wave * 16 + half * 8 + j) * SLD + n] = acc[j];
    }
  }
  __syncthreads();

  // V^T into Ks (all threads), softmax S -> P into Qs (first 128 threads)
  {
    const float4* v4 = (const float4*)(v + (size_t)b * T * d);
    for (int i = tid; i < T * d / 4; i += blockDim.x) {
      int s = i / (d / 4), c = (i % (d / 4)) * 4;
      float4 xv = v4[i];
      Ks[(c + 0) * ALD + s] = (_Float16)xv.x;
      Ks[(c + 1) * ALD + s] = (_Float16)xv.y;
      Ks[(c + 2) * ALD + s] = (_Float16)xv.z;
      Ks[(c + 3) * ALD + s] = (_Float16)xv.w;
    }
  }
  if (tid < T) {
    float* Sr = S + tid * SLD;
    float mx = Sr[0];
    for (int s = 1; s < T; ++s) mx = fmaxf(mx, Sr[s]);
    float sum = 0.0f;
    for (int s = 0; s < T; ++s) { float e = __expf(Sr[s] - mx); Sr[s] = e; sum += e; }
    float inv = 1.0f / sum;
    _Float16* Pr = Qs + tid * ALD;
    for (int s = 0; s < T; ++s) Pr[s] = (_Float16)(Sr[s] * inv);
  }
  __syncthreads();

  // O = P @ V : B fragment reads V^T rows (contiguous), A fragments hoisted.
  {
    v16h afp[4];
#pragma unroll
    for (int kk = 0; kk < 4; ++kk) {
      int k0 = kk * 32;
      v8h alo = *(const v8h*)(Qs + mrow * ALD + k0 + half * 8);
      v8h ahi = *(const v8h*)(Qs + mrow * ALD + k0 + 16 + half * 8);
      afp[kk] = __builtin_shufflevector(alo, ahi, 0,1,2,3,4,5,6,7,8,9,10,11,12,13,14,15);
    }
    for (int tn = 0; tn < d / 16; ++tn) {
      int n = tn * 16 + (lane & 15);
      v8f acc = {};
#pragma unroll
      for (int kk = 0; kk < 4; ++kk) {
        int k0 = kk * 32;
        v8h blo = *(const v8h*)(Ks + n * ALD + k0 + half * 8);
        v8h bhi = *(const v8h*)(Ks + n * ALD + k0 + 16 + half * 8);
        v16h bf = __builtin_shufflevector(blo, bhi, 0,1,2,3,4,5,6,7,8,9,10,11,12,13,14,15);
        acc = WMMA_F16(afp[kk], bf, acc);
      }
#pragma unroll
      for (int j = 0; j < 8; ++j)
        out[(size_t)b * out_bstride + (size_t)(wave * 16 + half * 8 + j) * out_rs + n] = acc[j];
    }
  }
}

// ---------------------------------------------------------------------------
// Head: out[b] = fc2( leaky_relu( (h0+h1) @ fc1^T + b1, 0.01 ) ) + b2
// ---------------------------------------------------------------------------
__global__ void k_final(const float* __restrict__ h0, const float* __restrict__ h1,
                        const float* __restrict__ W1, const float* __restrict__ b1,
                        const float* __restrict__ W2, const float* __restrict__ b2,
                        float* __restrict__ out, int B, int H) {
  int b = blockIdx.x * blockDim.x + threadIdx.x;
  if (b >= B) return;
  float hr[32];
  for (int j = 0; j < 32; ++j) {
    float s = b1[j];
    for (int kk = 0; kk < H; ++kk)
      s += (h0[b * H + kk] + h1[b * H + kk]) * W1[j * H + kk];
    hr[j] = s > 0.0f ? s : 0.01f * s;
  }
  float o = b2[0];
  for (int j = 0; j < 32; ++j) o += hr[j] * W2[j];
  out[b] = o;
}

// ---------------------------------------------------------------------------
// Host orchestration.
// Input leaf order (depth-first dict insertion order of setup_inputs()):
//  0 x(128,256,409)  1 state0(256,128)  2 state1(256,128)
//  gru_l0: 3 Wih 4 Whh 5 bih 6 bhh    gru_l1: 7..10
//  gru_a0: 11..14   gru_a1: 15..18    gru_v0: 19..22   gru_v1: 23..26
//  proj_l fwd 27..30 bwd 31..34 | proj_a fwd 35..38 bwd 39..42 | proj_v fwd 43..46 bwd 47..50
//  lstm fwd 51..54 bwd 55..58
//  jcaf l(Wq,Wk,Wv) 59..61  a 62..64  v 65..67
//  rgn0 68..71  rgn1 72..75  fc1_W 76 fc1_b 77 fc2_W 78 fc2_b 79
// ---------------------------------------------------------------------------
extern "C" void kernel_launch(void* const* d_in, const int* in_sizes, int n_in,
                              void* d_out, int out_size, void* d_ws, size_t ws_size,
                              hipStream_t stream) {
  (void)in_sizes; (void)n_in; (void)out_size; (void)ws_size;
  const int T = 128, B = 256, M = T * B;   // M = 32768
  const float* x = (const float*)d_in[0];
  auto F = [&](int i) { return (const float*)d_in[i]; };

  char* base = (char*)d_ws;
  size_t off = 0;
  auto take = [&](size_t bytes) -> void* {
    void* p = base + off;
    off = (off + bytes + 255) & ~(size_t)255;
    return p;
  };

  // cvt: weight (rows x cols) f32 -> (rows x colsp) f16, zero-padded
  auto cvt = [&](int idx, int rows, int cols, int colsp) -> const _Float16* {
    _Float16* dst = (_Float16*)take((size_t)rows * colsp * sizeof(_Float16));
    int n = rows * colsp;
    k_cvt_pad<<<dim3((n + 255) / 256), dim3(256), 0, stream>>>(F(idx), dst, rows, cols, colsp);
    return dst;
  };
  // cvtT: weight (K x N) f32 -> (N x K) f16 (K already mult of 32)
  auto cvtT = [&](int idx, int rows, int cols) -> const _Float16* {
    _Float16* dst = (_Float16*)take((size_t)rows * cols * sizeof(_Float16));
    int n = rows * cols;
    k_cvtT<<<dim3((n + 255) / 256), dim3(256), 0, stream>>>(F(idx), dst, rows, cols);
    return dst;
  };

  // ---- weights -> f16 (N x Kp row-major) ----
  const _Float16* W_l0_ih = cvt(3, 384, 300, 320);
  const _Float16* W_l0_hh = cvt(4, 384, 128, 128);
  const _Float16* W_l1_ih = cvt(7, 384, 128, 128);
  const _Float16* W_l1_hh = cvt(8, 384, 128, 128);
  const _Float16* W_a0_ih = cvt(11, 192, 74, 96);
  const _Float16* W_a0_hh = cvt(12, 192, 64, 64);
  const _Float16* W_a1_ih = cvt(15, 192, 64, 64);
  const _Float16* W_a1_hh = cvt(16, 192, 64, 64);
  const _Float16* W_v0_ih = cvt(19, 192, 35, 64);
  const _Float16* W_v0_hh = cvt(20, 192, 64, 64);
  const _Float16* W_v1_ih = cvt(23, 192, 64, 64);
  const _Float16* W_v1_hh = cvt(24, 192, 64, 64);
  const _Float16* Plf_ih = cvt(27, 192, 128, 128);
  const _Float16* Plf_hh = cvt(28, 192, 64, 64);
  const _Float16* Plb_ih = cvt(31, 192, 128, 128);
  const _Float16* Plb_hh = cvt(32, 192, 64, 64);
  const _Float16* Paf_ih = cvt(35, 192, 64, 64);
  const _Float16* Paf_hh = cvt(36, 192, 64, 64);
  const _Float16* Pab_ih = cvt(39, 192, 64, 64);
  const _Float16* Pab_hh = cvt(40, 192, 64, 64);
  const _Float16* Pvf_ih = cvt(43, 192, 64, 64);
  const _Float16* Pvf_hh = cvt(44, 192, 64, 64);
  const _Float16* Pvb_ih = cvt(47, 192, 64, 64);
  const _Float16* Pvb_hh = cvt(48, 192, 64, 64);
  const _Float16* Lf_ih = cvt(51, 512, 384, 384);
  const _Float16* Lf_hh = cvt(52, 512, 128, 128);
  const _Float16* Lb_ih = cvt(55, 512, 384, 384);
  const _Float16* Lb_hh = cvt(56, 512, 128, 128);
  const _Float16* Wq_l = cvtT(59, 128, 128);
  const _Float16* Wk_l = cvtT(60, 384, 128);
  const _Float16* Wv_l = cvtT(61, 384, 128);
  const _Float16* Wq_a = cvtT(62, 128, 128);
  const _Float16* Wk_a = cvtT(63, 384, 128);
  const _Float16* Wv_a = cvtT(64, 384, 128);
  const _Float16* Wq_v = cvtT(65, 128, 128);
  const _Float16* Wk_v = cvtT(66, 384, 128);
  const _Float16* Wv_v = cvtT(67, 384, 128);
  const _Float16* R0_ih = cvt(68, 384, 256, 256);
  const _Float16* R0_hh = cvt(69, 384, 128, 128);
  const _Float16* R1_ih = cvt(72, 384, 256, 256);
  const _Float16* R1_hh = cvt(73, 384, 128, 128);

  // ---- activation workspace ----
  float* xl_p = (float*)take((size_t)M * 320 * 4);   // packed x_l, K 300->320
  float* xa_p = (float*)take((size_t)M * 96 * 4);    // packed x_a, K 74->96
  float* xv_p = (float*)take((size_t)M * 64 * 4);    // packed x_v, K 35->64
  float* giA  = (float*)take((size_t)M * 512 * 4);
  float* giB  = (float*)take((size_t)M * 512 * 4);
  float* gh   = (float*)take((size_t)B * 512 * 4);
  float* htmp = (float*)take((size_t)B * 128 * 4);
  float* ctmp = (float*)take((size_t)B * 128 * 4);
  float* hr0  = (float*)take((size_t)B * 128 * 4);
  float* hr1  = (float*)take((size_t)B * 128 * 4);
  float* sl0  = (float*)take((size_t)M * 128 * 4);   // (T,B,128)
  float* sl1  = (float*)take((size_t)M * 128 * 4);
  float* sa0  = (float*)take((size_t)M * 64 * 4);    // (T,B,64)
  float* sa1  = (float*)take((size_t)M * 64 * 4);
  float* sv0  = (float*)take((size_t)M * 64 * 4);
  float* sv1  = (float*)take((size_t)M * 64 * 4);
  float* joint = (float*)take((size_t)M * 384 * 4);  // (B,T,384)
  float* qb   = (float*)take((size_t)M * 128 * 4);
  float* kb   = (float*)take((size_t)M * 128 * 4);
  float* vb   = (float*)take((size_t)M * 128 * 4);
  float* tstep = (float*)take((size_t)M * 384 * 4);  // (B,T,384)
  float* seq2 = (float*)take((size_t)M * 256 * 4);   // (T,B,256)

  // ---- pack ragged x slices ----
  k_pack<<<dim3((M * 320 + 255) / 256), dim3(256), 0, stream>>>(x + 0,   409, 300, xl_p, 320, M);
  k_pack<<<dim3((M * 96  + 255) / 256), dim3(256), 0, stream>>>(x + 300, 409, 74,  xa_p, 96,  M);
  k_pack<<<dim3((M * 64  + 255) / 256), dim3(256), 0, stream>>>(x + 374, 409, 35,  xv_p, 64,  M);

  // NT=4 (A-fragment reuse) for the big sequence-parallel GEMMs,
  // NT=1 for the small latency-bound recurrent GEMMs.
  auto gemm = [&](const float* A, int lda, const _Float16* W, const float* bias,
                  float* C, int ldc, int Mm, int Nn, int Kk) {
    if (Mm >= 4096 && (Nn % 64) == 0) {
      int groups = (Mm >> 4) * ((Nn >> 4) / 4);
      k_gemm_aw<4><<<dim3((groups + 7) / 8), dim3(256), 0, stream>>>(A, lda, W, bias, C, ldc, Mm, Nn, Kk);
    } else {
      int groups = (Mm >> 4) * (Nn >> 4);
      k_gemm_aw<1><<<dim3((groups + 7) / 8), dim3(256), 0, stream>>>(A, lda, W, bias, C, ldc, Mm, Nn, Kk);
    }
  };

  auto gru_scan = [&](const float* gi, size_t gi_ts, int gi_rs,
                      const _Float16* Whh, const float* bhh, float* h, int H,
                      float* out, size_t out_ts, long out_bs, int coff, bool rev) {
    int n = B * H;
    for (int s = 0; s < T; ++s) {
      int t = rev ? (T - 1 - s) : s;
      gemm(h, H, Whh, bhh, gh, 3 * H, B, 3 * H, H);
      float* op = out ? (out + (size_t)t * out_ts + coff) : nullptr;
      k_gru_gate<<<dim3((n + 255) / 256), dim3(256), 0, stream>>>(
          gi + (size_t)t * gi_ts, gi_rs, gh, h, op, out_bs, B, H);
    }
  };

  auto lstm_scan = [&](const float* gi, size_t gi_ts, int gi_rs,
                       const _Float16* Whh, const float* bhh, float* h, float* cs, int H,
                       float* out, size_t out_ts, long out_bs, int coff, bool rev) {
    int n = B * H;
    for (int s = 0; s < T; ++s) {
      int t = rev ? (T - 1 - s) : s;
      gemm(h, H, Whh, bhh, gh, 4 * H, B, 4 * H, H);
      k_lstm_gate<<<dim3((n + 255) / 256), dim3(256), 0, stream>>>(
          gi + (size_t)t * gi_ts, gi_rs, gh, h, cs, out + (size_t)t * out_ts + coff, out_bs, B, H);
    }
  };

  // ================= modality L =================
  gemm(xl_p, 320, W_l0_ih, F(5), giA, 384, M, 384, 320);
  hipMemsetAsync(htmp, 0, (size_t)B * 128 * 4, stream);
  gru_scan(giA, (size_t)B * 384, 384, W_l0_hh, F(6), htmp, 128, sl0, (size_t)B * 128, 128, 0, false);
  gemm(sl0, 128, W_l1_ih, F(9), giA, 384, M, 384, 128);
  hipMemsetAsync(htmp, 0, (size_t)B * 128 * 4, stream);
  gru_scan(giA, (size_t)B * 384, 384, W_l1_hh, F(10), htmp, 128, sl1, (size_t)B * 128, 128, 0, false);
  gemm(sl1, 128, Plf_ih, F(29), giA, 192, M, 192, 128);
  gemm(sl1, 128, Plb_ih, F(33), giB, 192, M, 192, 128);
  hipMemsetAsync(htmp, 0, (size_t)B * 64 * 4, stream);
  gru_scan(giA, (size_t)B * 192, 192, Plf_hh, F(30), htmp, 64, joint, 384, (long)T * 384, 0, false);
  hipMemsetAsync(htmp, 0, (size_t)B * 64 * 4, stream);
  gru_scan(giB, (size_t)B * 192, 192, Plb_hh, F(34), htmp, 64, joint, 384, (long)T * 384, 64, true);

  // ================= modality A =================
  gemm(xa_p, 96, W_a0_ih, F(13), giA, 192, M, 192, 96);
  hipMemsetAsync(htmp, 0, (size_t)B * 64 * 4, stream);
  gru_scan(giA, (size_t)B * 192, 192, W_a0_hh, F(14), htmp, 64, sa0, (size_t)B * 64, 64, 0, false);
  gemm(sa0, 64, W_a1_ih, F(17), giA, 192, M, 192, 64);
  hipMemsetAsync(htmp, 0, (size_t)B * 64 * 4, stream);
  gru_scan(giA, (size_t)B * 192, 192, W_a1_hh, F(18), htmp, 64, sa1, (size_t)B * 64, 64, 0, false);
  gemm(sa1, 64, Paf_ih, F(37), giA, 192, M, 192, 64);
  gemm(sa1, 64, Pab_ih, F(41), giB, 192, M, 192, 64);
  hipMemsetAsync(htmp, 0, (size_t)B * 64 * 4, stream);
  gru_scan(giA, (size_t)B * 192, 192, Paf_hh, F(38), htmp, 64, joint, 384, (long)T * 384, 128, false);
  hipMemsetAsync(htmp, 0, (size_t)B * 64 * 4, stream);
  gru_scan(giB, (size_t)B * 192, 192, Pab_hh, F(42), htmp, 64, joint, 384, (long)T * 384, 192, true);

  // ================= modality V =================
  gemm(xv_p, 64, W_v0_ih, F(21), giA, 192, M, 192, 64);
  hipMemsetAsync(htmp, 0, (size_t)B * 64 * 4, stream);
  gru_scan(giA, (size_t)B * 192, 192, W_v0_hh, F(22), htmp, 64, sv0, (size_t)B * 64, 64, 0, false);
  gemm(sv0, 64, W_v1_ih, F(25), giA, 192, M, 192, 64);
  hipMemsetAsync(htmp, 0, (size_t)B * 64 * 4, stream);
  gru_scan(giA, (size_t)B * 192, 192, W_v1_hh, F(26), htmp, 64, sv1, (size_t)B * 64, 64, 0, false);
  gemm(sv1, 64, Pvf_ih, F(45), giA, 192, M, 192, 64);
  gemm(sv1, 64, Pvb_ih, F(49), giB, 192, M, 192, 64);
  hipMemsetAsync(htmp, 0, (size_t)B * 64 * 4, stream);
  gru_scan(giA, (size_t)B * 192, 192, Pvf_hh, F(46), htmp, 64, joint, 384, (long)T * 384, 256, false);
  hipMemsetAsync(htmp, 0, (size_t)B * 64 * 4, stream);
  gru_scan(giB, (size_t)B * 192, 192, Pvb_hh, F(50), htmp, 64, joint, 384, (long)T * 384, 320, true);

  // ================= cross attention (JCAF) =================
  size_t attn_lds = (size_t)2 * 128 * ALD * sizeof(_Float16) + (size_t)128 * SLD * sizeof(float);
  const _Float16* Wq[3] = {Wq_l, Wq_a, Wq_v};
  const _Float16* Wk[3] = {Wk_l, Wk_a, Wk_v};
  const _Float16* Wv[3] = {Wv_l, Wv_a, Wv_v};
  for (int mmod = 0; mmod < 3; ++mmod) {
    int coff = mmod * 128;
    gemm(joint + coff, 384, Wq[mmod], nullptr, qb, 128, M, 128, 128);
    gemm(joint, 384, Wk[mmod], nullptr, kb, 128, M, 128, 384);
    gemm(joint, 384, Wv[mmod], nullptr, vb, 128, M, 128, 384);
    k_attn<<<dim3(B), dim3(256), attn_lds, stream>>>(qb, kb, vb, tstep + coff,
                                                     (long)T * 384, 384);
  }

  // ================= biLSTM fusion =================
  gemm(tstep, 384, Lf_ih, F(53), giA, 512, M, 512, 384);
  gemm(tstep, 384, Lb_ih, F(57), giB, 512, M, 512, 384);
  hipMemsetAsync(htmp, 0, (size_t)B * 128 * 4, stream);
  hipMemsetAsync(ctmp, 0, (size_t)B * 128 * 4, stream);
  // gi rows are (b,t): t-slice offset = t*512, per-b row stride = T*512
  lstm_scan(giA, 512, T * 512, Lf_hh, F(54), htmp, ctmp, 128, seq2, (size_t)B * 256, 256, 0, false);
  hipMemsetAsync(htmp, 0, (size_t)B * 128 * 4, stream);
  hipMemsetAsync(ctmp, 0, (size_t)B * 128 * 4, stream);
  lstm_scan(giB, 512, T * 512, Lb_hh, F(58), htmp, ctmp, 128, seq2, (size_t)B * 256, 256, 128, true);

  // ================= RGN scans =================
  gemm(seq2, 256, R0_ih, F(70), giA, 384, M, 384, 256);
  gemm(seq2, 256, R1_ih, F(74), giB, 384, M, 384, 256);
  hipMemcpyAsync(hr0, d_in[1], (size_t)B * 128 * 4, hipMemcpyDeviceToDevice, stream);
  hipMemcpyAsync(hr1, d_in[2], (size_t)B * 128 * 4, hipMemcpyDeviceToDevice, stream);
  gru_scan(giA, (size_t)B * 384, 384, R0_hh, F(71), hr0, 128, nullptr, 0, 0, 0, false);
  gru_scan(giB, (size_t)B * 384, 384, R1_hh, F(75), hr1, 128, nullptr, 0, 0, 0, true);

  // ================= head =================
  k_final<<<dim3(1), dim3(256), 0, stream>>>(hr0, hr1, F(76), F(77), F(78), F(79),
                                             (float*)d_out, B, 128);
}
